// GNNModel_1511828488915
// MI455X (gfx1250) — matrix operand order repaired and checked by reading, not compiled, not run on hardware
//
#include <hip/hip_runtime.h>
#include <stdint.h>

// ---------------------------------------------------------------------------
// Types for CDNA5 WMMA
// ---------------------------------------------------------------------------
typedef __attribute__((ext_vector_type(16))) __bf16 v16bf;
typedef __attribute__((ext_vector_type(8)))  float  v8f;

struct Frag32B { uint4 lo, hi; };   // 32 bytes == one v16bf

// fp32 -> bf16 (round to nearest even)
__device__ __forceinline__ unsigned short f2bf(float f) {
    unsigned u = __builtin_bit_cast(unsigned, f);
    u += 0x7FFFu + ((u >> 16) & 1u);
    return (unsigned short)(u >> 16);
}

// Load a 32B fragment from LDS as 2 x ds_load_b128
__device__ __forceinline__ v16bf load_frag(const unsigned short* p) {
    const uint4* q = reinterpret_cast<const uint4*>(p);
    Frag32B t;
    t.lo = q[0];
    t.hi = q[1];
    return __builtin_bit_cast(v16bf, t);
}

// Async copy 16B/lane global -> LDS (GLOBAL_LOAD_ASYNC_TO_LDS_B128, ASYNCcnt).
// ldsOff = addrspace(3) byte offset (low 32 bits of the generic pointer).
__device__ __forceinline__ void async_copy_b128(unsigned ldsOff, const void* g) {
    unsigned long long ga = (unsigned long long)(uintptr_t)g;
    asm volatile("global_load_async_to_lds_b128 %0, %1, off"
                 :: "v"(ldsOff), "v"(ga) : "memory");
}

__device__ __forceinline__ void wait_asynccnt0() {
#if __has_builtin(__builtin_amdgcn_s_wait_asynccnt)
    __builtin_amdgcn_s_wait_asynccnt(0);
#else
    asm volatile("s_wait_asynccnt 0" ::: "memory");
#endif
}

__device__ __forceinline__ void atom_add(float* p, float v) {
    unsafeAtomicAdd(p, v);   // global_atomic_add_f32 on gfx1250
}

// ---------------------------------------------------------------------------
// Degree / normalization kernels
// ---------------------------------------------------------------------------
__global__ void k_init_deg(float* __restrict__ deg, int n) {
    int i = blockIdx.x * blockDim.x + threadIdx.x;
    if (i < n) deg[i] = 1.0f;   // self-loop contributes 1 to every node
}

__global__ void k_deg_accum(const long long* __restrict__ dst,
                            float* __restrict__ deg, int E) {
    int i = blockIdx.x * blockDim.x + threadIdx.x;
    if (i < E) atom_add(&deg[(int)dst[i]], 1.0f);
}

__global__ void k_rsqrt(float* __restrict__ d, int n) {
    int i = blockIdx.x * blockDim.x + threadIdx.x;
    if (i < n) d[i] = rsqrtf(d[i]);   // deg >= 1 always
}

// ---------------------------------------------------------------------------
// Weight prepack: fp32 row-major [K][N]  ->  bf16, WMMA-B fragment order.
// Layout: Wp[kb][n][h*16 + e], row stride 40 halves (pad vs bank conflicts),
// where k = kb*32 + h*16 + e  (k order used here: lanes 0-15 K=0..15, 16-31 K=16..31).
// ---------------------------------------------------------------------------
__global__ void k_pack_w(const float* __restrict__ W, unsigned short* __restrict__ Wp,
                         int K, int N) {
    int i = blockIdx.x * blockDim.x + threadIdx.x;
    if (i >= K * N) return;
    int k = i / N, n = i % N;
    int kb = k >> 5, kr = k & 31;
    int h = kr >> 4, e = kr & 15;
    Wp[(size_t)kb * N * 40 + (size_t)n * 40 + h * 16 + e] = f2bf(W[i]);
}

// ---------------------------------------------------------------------------
// Double-buffered tiled GEMM with v_wmma_f32_16x16x32_bf16.
// C[M,N] = A[M,K] * Wp(prepacked B). A: fp32 streamed, converted to bf16
// fragment order in LDS. B: async global->LDS copy of prepacked weights.
// One barrier per k-step; stage(kb+1) overlaps compute(kb).
// Block = 256 threads = 8 waves; wave owns a 16 x 64 output strip.
// Requires BN == N and K % 32 == 0.
// ---------------------------------------------------------------------------
template<int BM, int BN>
__global__ __launch_bounds__(256) void gemm_wmma_bf16(
    const float* __restrict__ A, const unsigned short* __restrict__ Bp,
    float* __restrict__ C, int M, int N, int K)
{
    constexpr int WM = BM / 16;   // waves along M
    constexpr int WN = 8 / WM;    // wave groups along N, 64 cols each
    static_assert(WN * 64 == BN, "tile mismatch");

    // Two fragment-ordered tile buffers (row stride 40 halves = 80B).
    __shared__ __align__(16) unsigned short Af[2][BM * 40];
    __shared__ __align__(16) unsigned short Bf[2][BN * 40];

    const int tid   = threadIdx.x;
    const int wave  = tid >> 5;
    const int lane  = tid & 31;
    const int half  = lane >> 4;
    const int l16   = lane & 15;
    const int waveM = wave % WM;
    const int waveN = wave / WM;
    const int mBlk  = blockIdx.x * BM;
    const int m0    = mBlk + waveM * 16;
    const int n0    = waveN * 64;

    // ---- B stage: uniform, constexpr trip count async copies ----
    auto stageB = [&](int kb, int buf) {
        const unsigned short* bsrc = Bp + (size_t)kb * N * 40;
        constexpr int TOT  = BN * 5;        // uint4 slots (BN*80B / 16B)
        constexpr int FULL = TOT / 256;
        #pragma unroll
        for (int i = 0; i < FULL; ++i) {
            int idx = tid + i * 256;
            unsigned ldsOff = (unsigned)(uintptr_t)(const void*)&Bf[buf][idx * 8];
            async_copy_b128(ldsOff, bsrc + (size_t)idx * 8);
        }
        if constexpr (TOT % 256) {
            if (tid < TOT % 256) {
                int idx = tid + FULL * 256;
                unsigned ldsOff = (unsigned)(uintptr_t)(const void*)&Bf[buf][idx * 8];
                async_copy_b128(ldsOff, bsrc + (size_t)idx * 8);
            }
        }
    };

    // ---- A stage: 8 consecutive k of one row -> one packed ds_store_b128.
    // A layout: h = k>>4, e = k&15: an 8-aligned run of k
    // maps to one (h, e0) run of 8 halves.
    auto stageA = [&](int k0, int buf) {
        auto doChunk = [&](int ch) {
            int r  = ch >> 2;
            int kc = (ch & 3) * 8;
            int gm = mBlk + r;
            float v[8] = {};
            if (gm < M) {
                const float* ap = A + (size_t)gm * K + k0 + kc;
                if (k0 + 32 < K) __builtin_prefetch(ap + 32, 0, 1);  // next slab
                float4 f0 = *reinterpret_cast<const float4*>(ap);
                float4 f1 = *reinterpret_cast<const float4*>(ap + 4);
                v[0] = f0.x; v[1] = f0.y; v[2] = f0.z; v[3] = f0.w;
                v[4] = f1.x; v[5] = f1.y; v[6] = f1.z; v[7] = f1.w;
            }
            uint4 pk;
            pk.x = f2bf(v[0]) | ((unsigned)f2bf(v[1]) << 16);
            pk.y = f2bf(v[2]) | ((unsigned)f2bf(v[3]) << 16);
            pk.z = f2bf(v[4]) | ((unsigned)f2bf(v[5]) << 16);
            pk.w = f2bf(v[6]) | ((unsigned)f2bf(v[7]) << 16);
            int h  = kc >> 4;
            int e0 = kc & 8;             // 0 or 8
            *reinterpret_cast<uint4*>(&Af[buf][r * 40 + h * 16 + e0]) = pk;
        };
        constexpr int CH = BM * 4;
        if constexpr (CH >= 256) {
            #pragma unroll
            for (int i = 0; i < CH / 256; ++i) doChunk(tid + i * 256);
        } else {
            if (tid < CH) doChunk(tid);
        }
    };

    v8f acc[4] = {};
    const int nk = K >> 5;

    // Prologue: stage tile 0 into buffer 0.
    stageB(0, 0);
    stageA(0, 0);

    for (int kb = 0; kb < nk; ++kb) {
        const int cur = kb & 1;

        // Previous iteration's async copies (into buf cur) must be in LDS,
        // then make all waves' stores visible.
        wait_asynccnt0();
        __syncthreads();

        // Overlap: stage next tile into the other buffer during compute.
        if (kb + 1 < nk) {
            stageB(kb + 1, cur ^ 1);
            stageA((kb + 1) * 32, cur ^ 1);
        }

        // ---- Compute: 2 b128 loads per fragment, 4 WMMAs per wave ----
        v16bf a = load_frag(&Af[cur][(waveM * 16 + l16) * 40 + half * 16]);
        #pragma unroll
        for (int nt = 0; nt < 4; ++nt) {
            v16bf b = load_frag(&Bf[cur][(n0 + nt * 16 + l16) * 40 + half * 16]);
            acc[nt] = __builtin_amdgcn_wmma_f32_16x16x32_bf16(
                false, a, false, b, (short)0, acc[nt], false, false);
        }
    }

    // C/D layout: VGPR r -> row m0 + 8*half + r, col = n-tile + (lane&15)
    #pragma unroll
    for (int nt = 0; nt < 4; ++nt) {
        #pragma unroll
        for (int r = 0; r < 8; ++r) {
            int m = m0 + half * 8 + r;
            int n = n0 + nt * 16 + l16;
            if (m < M) C[(size_t)m * N + n] = acc[nt][r];
        }
    }
}

// ---------------------------------------------------------------------------
// Edge aggregation: out[dst] += dinv[src]*dinv[dst] * h[src]  (segment sum)
// ---------------------------------------------------------------------------
template<int C>
__global__ __launch_bounds__(256) void k_edge_agg(
    const long long* __restrict__ src, const long long* __restrict__ dst,
    const float* __restrict__ dinv, const float* __restrict__ h,
    float* __restrict__ out, int E)
{
    constexpr int LPE = C / 4;        // lanes per edge
    constexpr int EPB = 256 / LPE;    // edges per block
    const int le = threadIdx.x / LPE;
    const int f  = threadIdx.x % LPE;
    const int e  = blockIdx.x * EPB + le;
    if (e >= E) return;

    const int s = (int)src[e];
    const int d = (int)dst[e];
    const float w = dinv[s] * dinv[d];

    const float4 hv = *reinterpret_cast<const float4*>(h + (size_t)s * C + f * 4);
    float* op = out + (size_t)d * C + f * 4;
    atom_add(op + 0, w * hv.x);
    atom_add(op + 1, w * hv.y);
    atom_add(op + 2, w * hv.z);
    atom_add(op + 3, w * hv.w);
}

// ---------------------------------------------------------------------------
// Epilogue: io[n,c] = (io[n,c] + dinv[n]^2 * hself[n,c] + bias[c]) [relu?]
// ---------------------------------------------------------------------------
template<int C, bool RELU>
__global__ __launch_bounds__(256) void k_finish(
    const float* __restrict__ hself, const float* __restrict__ bias,
    const float* __restrict__ dinv, float* __restrict__ io, size_t total)
{
    size_t i = (size_t)blockIdx.x * blockDim.x + threadIdx.x;
    if (i >= total) return;
    int    c = (int)(i % C);
    size_t n = i / C;
    float dv = dinv[n];
    float v  = io[i] + dv * dv * hself[i] + bias[c];
    io[i] = RELU ? fmaxf(v, 0.0f) : v;
}

// ---------------------------------------------------------------------------
// Host launcher
// ---------------------------------------------------------------------------
extern "C" void kernel_launch(void* const* d_in, const int* in_sizes, int n_in,
                              void* d_out, int out_size, void* d_ws, size_t ws_size,
                              hipStream_t stream) {
    (void)n_in; (void)ws_size;
    const float*     x   = (const float*)d_in[0];
    const long long* ei  = (const long long*)d_in[1];  // int64 edge_index [2,E]
    const float*     W1  = (const float*)d_in[2];
    const float*     b1  = (const float*)d_in[3];
    const float*     W2  = (const float*)d_in[4];
    const float*     b2  = (const float*)d_in[5];
    float*           out = (float*)d_out;

    constexpr int IN_C = 512, HID = 256, OUT = 128;
    const int N = in_sizes[0] / IN_C;       // 50000
    const int E = in_sizes[1] / 2;          // 800000
    const long long* srcI = ei;
    const long long* dstI = ei + E;

    // Workspace layout (floats): dinv | h1 (reused as t2) | agg1 | Wp1 | Wp2
    float* ws   = (float*)d_ws;
    size_t off  = ((size_t)N + 255) & ~(size_t)255;
    float* dinv = ws;
    float* h1   = ws + off;             off += (size_t)N * HID;
    float* agg1 = ws + off;             off += (size_t)N * HID;
    unsigned short* Wp1 = (unsigned short*)(ws + off);
    size_t wp1_halves = ((size_t)IN_C / 32) * HID * 40;            // 16*256*40
    unsigned short* Wp2 = Wp1 + ((wp1_halves + 7) & ~(size_t)7);
    float* t2   = h1;                   // h1 dead after k_finish<HID>; reuse

    // Zero the atomic accumulation targets
    hipMemsetAsync(agg1, 0, (size_t)N * HID * sizeof(float), stream);
    hipMemsetAsync(out,  0, (size_t)out_size * sizeof(float), stream);

    // Prepack weights into bf16 fragment order (once, L2 resident after)
    k_pack_w<<<(IN_C * HID + 255) / 256, 256, 0, stream>>>(W1, Wp1, IN_C, HID);
    k_pack_w<<<(HID * OUT + 255) / 256, 256, 0, stream>>>(W2, Wp2, HID, OUT);

    // Symmetric normalization: deg (incl. self loop) -> dinv = rsqrt(deg)
    k_init_deg <<<(N + 255) / 256, 256, 0, stream>>>(dinv, N);
    k_deg_accum<<<(E + 255) / 256, 256, 0, stream>>>(dstI, dinv, E);
    k_rsqrt    <<<(N + 255) / 256, 256, 0, stream>>>(dinv, N);

    // Layer 1: h1 = x @ W1 (WMMA), aggregate, self+bias+ReLU -> agg1
    gemm_wmma_bf16<32, 256><<<(N + 31) / 32, 256, 0, stream>>>(x, Wp1, h1, N, HID, IN_C);
    k_edge_agg<HID><<<(E + 3) / 4, 256, 0, stream>>>(srcI, dstI, dinv, h1, agg1, E);
    k_finish<HID, true><<<(unsigned)(((size_t)N * HID + 255) / 256), 256, 0, stream>>>(
        h1, b1, dinv, agg1, (size_t)N * HID);

    // Layer 2: t2 = relu_h @ W2 (WMMA), aggregate into out, self+bias
    gemm_wmma_bf16<64, 128><<<(N + 63) / 64, 256, 0, stream>>>(agg1, Wp2, t2, N, OUT, HID);
    k_edge_agg<OUT><<<(E + 7) / 8, 256, 0, stream>>>(srcI, dstI, dinv, t2, out, E);
    k_finish<OUT, false><<<(unsigned)(((size_t)N * OUT + 255) / 256), 256, 0, stream>>>(
        t2, b2, dinv, out, (size_t)N * OUT);
}
